// SchNetInteractionTriple_80590766342347
// MI455X (gfx1250) — matrix operand order repaired and checked by reading, not compiled, not run on hardware
//
#include <hip/hip_runtime.h>

#define NF      128
#define LDS_S   136      // LDS row stride in halves (bank-conflict-free, 16B aligned)
#define NATOMS  2048
#define NT      128
#define NN      64
#define TIN     100
#define NSB     25

typedef __attribute__((ext_vector_type(8)))  _Float16 v8h;
typedef __attribute__((ext_vector_type(16))) _Float16 v16h;
typedef __attribute__((ext_vector_type(8)))  float    v8f;

__device__ __forceinline__ v8f vzero8() {
    v8f z;
#pragma unroll
    for (int i = 0; i < 8; i++) z[i] = 0.0f;
    return z;
}

// shifted softplus: softplus(x) - ln2, numerically stable
__device__ __forceinline__ float sspf(float x) {
    float ax = fabsf(x);
    return fmaxf(x, 0.0f) + log1pf(__expf(-ax)) - 0.69314718055994531f;
}

__device__ __forceinline__ float ccutf(float r) {
    const float rc = 5.0f;
    float v = 0.5f * (__cosf(3.14159265358979323f * r / rc) + 1.0f);
    return (r < rc) ? v : 0.0f;
}

// A fragment 16x32 f16 (ISA 7.12.2): lane l(0..15),hi: h[0..7]=K(k0+8hi+i), h[8..15]=K(k0+16+8hi+i)
__device__ __forceinline__ v16h load_fragA(const _Float16* base, int rowBase, int k0, int lane) {
    const int l = lane & 15, hi = (lane >> 4) & 1;
    const _Float16* p = base + (rowBase + l) * LDS_S + k0 + 8 * hi;
    v8h lo = *(const v8h*)p;
    v8h hh = *(const v8h*)(p + 16);
    v16h r;
#pragma unroll
    for (int i = 0; i < 8; i++) { r[i] = lo[i]; r[i + 8] = hh[i]; }
    return r;
}

// B fragment 32x16 f16: lane group holds N=l, h[0..7]=K(k0+8hi+i), h[8..15]=K(k0+16+8hi+i).
// baseT is B^T stored [N][K] row-major so each group of 8 K's is contiguous.
__device__ __forceinline__ v16h load_fragB(const _Float16* baseT, int colBase, int k0, int lane) {
    const int l = lane & 15, hi = (lane >> 4) & 1;
    const _Float16* p = baseT + (colBase + l) * LDS_S + k0 + 8 * hi;
    v8h lo = *(const v8h*)p;
    v8h hh = *(const v8h*)(p + 16);
    v16h r;
#pragma unroll
    for (int i = 0; i < 8; i++) { r[i] = lo[i]; r[i + 8] = hh[i]; }
    return r;
}

__device__ __forceinline__ v8f wmma16(v16h a, v16h b, v8f c) {
    return __builtin_amdgcn_wmma_f32_16x16x32_f16(false, a, false, b, (short)0, c, false, false);
}

// ---------------------------------------------------------------------------
// Kernel 0: y = x @ Win   (2048x128)(128x128), f16 WMMA, f32 accumulate
// ---------------------------------------------------------------------------
__global__ __launch_bounds__(256) void in2f_kernel(const float* __restrict__ x,
                                                   const float* __restrict__ Win,
                                                   float* __restrict__ y) {
    extern __shared__ char smem[];
    _Float16* sA  = (_Float16*)smem;              // 128 x LDS_S
    _Float16* sBT = sA + 128 * LDS_S;             // 128 x LDS_S
    const int tid = threadIdx.x;
    const int rowBlk = blockIdx.x * 128;

    for (int idx = tid; idx < 128 * 128; idx += 256) {
        int r = idx >> 7, k = idx & 127;
        sA[r * LDS_S + k] = (_Float16)x[(rowBlk + r) * NF + k];
    }
    for (int idx = tid; idx < 128 * 128; idx += 256) {
        int k = idx >> 7, n = idx & 127;
        sBT[n * LDS_S + k] = (_Float16)Win[k * NF + n];
    }
    __syncthreads();

    const int wave = tid >> 5, lane = tid & 31;
    const int l = lane & 15, hi = (lane >> 4) & 1;
    const int rowBase = wave * 16;

    v8f acc[8];
#pragma unroll
    for (int nt = 0; nt < 8; nt++) acc[nt] = vzero8();
#pragma unroll
    for (int kb = 0; kb < 4; kb++) {
        v16h aF = load_fragA(sA, rowBase, kb * 32, lane);
#pragma unroll
        for (int nt = 0; nt < 8; nt++) {
            v16h bF = load_fragB(sBT, nt * 16, kb * 32, lane);
            acc[nt] = wmma16(aF, bF, acc[nt]);
        }
    }
#pragma unroll
    for (int nt = 0; nt < 8; nt++) {
#pragma unroll
        for (int v = 0; v < 8; v++) {
            int row = rowBlk + rowBase + v + 8 * hi;
            int col = nt * 16 + l;
            y[row * NF + col] = acc[nt][v];
        }
    }
}

// ---------------------------------------------------------------------------
// Kernel 1: fused per-atom interaction (one workgroup per atom)
// ---------------------------------------------------------------------------
__global__ __launch_bounds__(256) void interaction_kernel(
    const float* __restrict__ r_double, const float* __restrict__ r_ij,
    const float* __restrict__ r_jk, const float* __restrict__ nmask,
    const float* __restrict__ tmask, const float* __restrict__ d_ijk,
    const float* __restrict__ f_double,
    const float* __restrict__ Wd1, const float* __restrict__ bd1,
    const float* __restrict__ Wd2, const float* __restrict__ bd2,
    const float* __restrict__ Wt1, const float* __restrict__ bt1,
    const float* __restrict__ Wt2, const float* __restrict__ bt2,
    const float* __restrict__ Wout, const float* __restrict__ bout,
    const float* __restrict__ Wdense, const float* __restrict__ bdense,
    const int* __restrict__ nbr, const int* __restrict__ nbrJ,
    const int* __restrict__ nbrK,
    const float* __restrict__ y, float* __restrict__ out) {
    const int a = blockIdx.x;
    const int tid = threadIdx.x;

    extern __shared__ char smem[];
    _Float16* sA    = (_Float16*)smem;                    // 128 x LDS_S (34816 B)
    _Float16* sBT1  = sA + 128 * LDS_S;                   // 34816 B
    _Float16* sBT2  = sBT1 + 128 * LDS_S;                 // 34816 B
    float*    partial = (float*)(sBT2 + 128 * LDS_S);     // 8 x 128 (4096 B)
    float*    cutT  = partial + 8 * 128;                  // 128
    int*      jI    = (int*)(cutT + 128);                 // 128
    int*      kI    = jI + 128;                           // 128
    float*    cutD  = (float*)(kI + 128);                 // 64
    int*      nI    = (int*)(cutD + 64);                  // 64
    float*    vtmp  = (float*)(nI + 64);                  // 128

    // ---- stage triple-branch operands (f32 -> f16, zero-pad K to 128) ----
    for (int idx = tid; idx < 128 * 128; idx += 256) {
        int t = idx >> 7, k = idx & 127;
        float v = (k < TIN) ? d_ijk[(a * NT + t) * TIN + k] : 0.0f;
        sA[t * LDS_S + k] = (_Float16)v;
    }
    for (int idx = tid; idx < 128 * 128; idx += 256) {
        int k = idx >> 7, n = idx & 127;
        float v = (k < TIN) ? Wt1[k * NF + n] : 0.0f;
        sBT1[n * LDS_S + k] = (_Float16)v;
    }
    for (int idx = tid; idx < 128 * 128; idx += 256) {
        int k = idx >> 7, n = idx & 127;
        sBT2[n * LDS_S + k] = (_Float16)Wt2[k * NF + n];
    }
    if (tid < NT) {
        int t = tid;
        cutT[t] = ccutf(r_ij[a * NT + t]) * ccutf(r_jk[a * NT + t]) * tmask[a * NT + t];
        jI[t] = nbrJ[a * NT + t];
        kI[t] = nbrK[a * NT + t];
    }
    for (int idx = tid; idx < 8 * 128; idx += 256) partial[idx] = 0.0f;
    __syncthreads();

    const int wave = tid >> 5, lane = tid & 31;
    const int l = lane & 15, hi = (lane >> 4) & 1;
    const int rowBase = wave * 16;

    // ---- triple GEMM1: H1 = ssp(d_ijk @ Wt1 + bt1) ----
    v8f acc[8];
#pragma unroll
    for (int nt = 0; nt < 8; nt++) acc[nt] = vzero8();
#pragma unroll
    for (int kb = 0; kb < 4; kb++) {
        v16h aF = load_fragA(sA, rowBase, kb * 32, lane);
#pragma unroll
        for (int nt = 0; nt < 8; nt++) {
            v16h bF = load_fragB(sBT1, nt * 16, kb * 32, lane);
            acc[nt] = wmma16(aF, bF, acc[nt]);
        }
    }
    __syncthreads();   // all GEMM1 reads of sA done
#pragma unroll
    for (int nt = 0; nt < 8; nt++) {
        int col = nt * 16 + l;
        float b = bt1[col];
#pragma unroll
        for (int v = 0; v < 8; v++) {
            int row = rowBase + v + 8 * hi;
            sA[row * LDS_S + col] = (_Float16)sspf(acc[nt][v] + b);
        }
    }
    __syncthreads();

    // ---- triple GEMM2: W_pre = H1 @ Wt2 ----
#pragma unroll
    for (int nt = 0; nt < 8; nt++) acc[nt] = vzero8();
#pragma unroll
    for (int kb = 0; kb < 4; kb++) {
        v16h aF = load_fragA(sA, rowBase, kb * 32, lane);
#pragma unroll
        for (int nt = 0; nt < 8; nt++) {
            v16h bF = load_fragB(sBT2, nt * 16, kb * 32, lane);
            acc[nt] = wmma16(aF, bF, acc[nt]);
        }
    }
    // ---- triple epilogue: yt += ssp(.+bt2)*cut * y[j]*y[k], reduce over t ----
#pragma unroll
    for (int nt = 0; nt < 8; nt++) {
        int col = nt * 16 + l;
        float b = bt2[col];
        float sum = 0.0f;
#pragma unroll
        for (int v = 0; v < 8; v++) {
            int t = rowBase + v + 8 * hi;
            float w = sspf(acc[nt][v] + b) * cutT[t];
            float yj = y[jI[t] * NF + col];
            float yk = y[kI[t] * NF + col];
            sum += w * yj * yk;
        }
        sum += __shfl_xor(sum, 16, 32);           // fold the two lane halves
        if (hi == 0) partial[wave * 128 + col] += sum;
    }
    __syncthreads();   // GEMM2 reads + partial writes done; safe to restage

    // ---- stage double-branch operands ----
    for (int idx = tid; idx < NN * 32; idx += 256) {
        int n = idx >> 5, k = idx & 31;
        float v = (k < NSB) ? f_double[(a * NN + n) * NSB + k] : 0.0f;
        sA[n * LDS_S + k] = (_Float16)v;
    }
    for (int idx = tid; idx < 128 * 32; idx += 256) {
        int k = idx >> 7, n = idx & 127;       // k in 0..31
        float v = (k < NSB) ? Wd1[k * NF + n] : 0.0f;
        sBT1[n * LDS_S + k] = (_Float16)v;
    }
    for (int idx = tid; idx < 128 * 128; idx += 256) {
        int k = idx >> 7, n = idx & 127;
        sBT2[n * LDS_S + k] = (_Float16)Wd2[k * NF + n];
    }
    if (tid < NN) {
        cutD[tid] = ccutf(r_double[a * NN + tid]) * nmask[a * NN + tid];
        nI[tid] = nbr[a * NN + tid];
    }
    __syncthreads();

    // ---- double GEMM1 (64 rows -> waves 0..3): ssp(f_double @ Wd1 + bd1) ----
#pragma unroll
    for (int nt = 0; nt < 8; nt++) acc[nt] = vzero8();
    if (wave < 4) {
        v16h aF = load_fragA(sA, rowBase, 0, lane);
#pragma unroll
        for (int nt = 0; nt < 8; nt++) {
            v16h bF = load_fragB(sBT1, nt * 16, 0, lane);
            acc[nt] = wmma16(aF, bF, acc[nt]);
        }
    }
    __syncthreads();
    if (wave < 4) {
#pragma unroll
        for (int nt = 0; nt < 8; nt++) {
            int col = nt * 16 + l;
            float b = bd1[col];
#pragma unroll
            for (int v = 0; v < 8; v++) {
                int row = rowBase + v + 8 * hi;
                sA[row * LDS_S + col] = (_Float16)sspf(acc[nt][v] + b);
            }
        }
    }
    __syncthreads();

    // ---- double GEMM2 + epilogue ----
    if (wave < 4) {
#pragma unroll
        for (int nt = 0; nt < 8; nt++) acc[nt] = vzero8();
#pragma unroll
        for (int kb = 0; kb < 4; kb++) {
            v16h aF = load_fragA(sA, rowBase, kb * 32, lane);
#pragma unroll
            for (int nt = 0; nt < 8; nt++) {
                v16h bF = load_fragB(sBT2, nt * 16, kb * 32, lane);
                acc[nt] = wmma16(aF, bF, acc[nt]);
            }
        }
#pragma unroll
        for (int nt = 0; nt < 8; nt++) {
            int col = nt * 16 + l;
            float b = bd2[col];
            float sum = 0.0f;
#pragma unroll
            for (int v = 0; v < 8; v++) {
                int n = rowBase + v + 8 * hi;
                float w = sspf(acc[nt][v] + b) * cutD[n];
                sum += w * y[nI[n] * NF + col];
            }
            sum += __shfl_xor(sum, 16, 32);
            if (hi == 0) partial[wave * 128 + col] += sum;
        }
    }
    __syncthreads();

    // ---- reduce partials: s = yd + yt ----
    if (tid < 128) {
        float s = 0.0f;
#pragma unroll
        for (int wv = 0; wv < 8; wv++) s += partial[wv * 128 + tid];
        cutT[tid] = s;   // reuse as s-vector
    }
    __syncthreads();
    // ---- v = ssp(s @ Wout + bout) ----
    if (tid < 128) {
        float av = bout[tid];
        for (int k = 0; k < 128; k++) av += cutT[k] * Wout[k * NF + tid];
        vtmp[tid] = sspf(av);
    }
    __syncthreads();
    // ---- out = v @ Wdense + bdense ----
    if (tid < 128) {
        float o = bdense[tid];
        for (int k = 0; k < 128; k++) o += vtmp[k] * Wdense[k * NF + tid];
        out[a * NF + tid] = o;
    }
}

extern "C" void kernel_launch(void* const* d_in, const int* in_sizes, int n_in,
                              void* d_out, int out_size, void* d_ws, size_t ws_size,
                              hipStream_t stream) {
    const float* x        = (const float*)d_in[0];
    const float* r_double = (const float*)d_in[1];
    const float* r_ij     = (const float*)d_in[2];
    const float* r_jk     = (const float*)d_in[3];
    const float* nmask    = (const float*)d_in[4];
    const float* tmask    = (const float*)d_in[5];
    const float* d_ijk    = (const float*)d_in[6];
    const float* f_double = (const float*)d_in[7];
    const float* Wd1      = (const float*)d_in[8];
    const float* bd1      = (const float*)d_in[9];
    const float* Wd2      = (const float*)d_in[10];
    const float* bd2      = (const float*)d_in[11];
    const float* Wt1      = (const float*)d_in[12];
    const float* bt1      = (const float*)d_in[13];
    const float* Wt2      = (const float*)d_in[14];
    const float* bt2      = (const float*)d_in[15];
    const float* Win      = (const float*)d_in[16];
    const float* Wout     = (const float*)d_in[17];
    const float* bout     = (const float*)d_in[18];
    const float* Wdense   = (const float*)d_in[19];
    const float* bdense   = (const float*)d_in[20];
    const int*   nbr      = (const int*)d_in[21];
    const int*   nbrJ     = (const int*)d_in[22];
    const int*   nbrK     = (const int*)d_in[23];
    float* out = (float*)d_out;

    float* y = (float*)d_ws;   // 2048 * 128 f32 = 1 MB

    const size_t sh0 = 2u * 128u * LDS_S * sizeof(_Float16);
    in2f_kernel<<<dim3(NATOMS / 128), dim3(256), sh0, stream>>>(x, Win, y);

    const size_t sh1 = 3u * 128u * LDS_S * sizeof(_Float16)   // sA, sBT1, sBT2
                     + (8 * 128 + 128 + 64 + 128) * sizeof(float)  // partial,cutT,cutD,vtmp
                     + (128 + 128 + 64) * sizeof(int);             // jI,kI,nI
    interaction_kernel<<<dim3(NATOMS), dim3(256), sh1, stream>>>(
        r_double, r_ij, r_jk, nmask, tmask, d_ijk, f_double,
        Wd1, bd1, Wd2, bd2, Wt1, bt1, Wt2, bt2,
        Wout, bout, Wdense, bdense, nbr, nbrJ, nbrK, y, out);
}
